// DPNET_73169062855374
// MI455X (gfx1250) — compile-verified, matrix operand order checked by reading, not run on hardware
//
#include <hip/hip_runtime.h>
#include <cstdint>

#define V_ 3
#define N_ 4096
#define D_ 512
#define H_ 128
#define C_ 5

// ---------------- WMMA types / helpers (gfx1250, wave32) ----------------
typedef __attribute__((ext_vector_type(16))) __bf16 v16bf;
typedef __attribute__((ext_vector_type(8)))  float  v8f;
typedef __attribute__((ext_vector_type(4)))  unsigned int u32x4;
typedef __attribute__((ext_vector_type(8)))  int i32x8;
typedef __attribute__((ext_vector_type(4)))  int i32x4;

union BfVec { uint4 q[2]; v16bf v; };

// TDM available?
#if defined(__has_builtin)
# if __has_builtin(__builtin_amdgcn_tensor_load_to_lds)
#  define USE_TDM 1
# endif
#endif
#ifndef USE_TDM
# define USE_TDM 0
#endif

__device__ inline unsigned short f2bf(float f) {
    unsigned int u = __builtin_bit_cast(unsigned int, f);
    u = (u + 0x7FFFu + ((u >> 16) & 1u)) >> 16;   // round-to-nearest-even
    return (unsigned short)u;
}

// Load 16 bf16 elements in the 16-bit A/B WMMA striping:
// elements [0..7] = K base..base+7, elements [8..15] = K base+16..base+23
__device__ inline BfVec load16(const unsigned short* p) {
    BfVec r;
    r.q[0] = *(const uint4*)(p);
    r.q[1] = *(const uint4*)(p + 16);
    return r;
}

__device__ inline v8f wmma_bf16(const BfVec& a, const BfVec& b, v8f c) {
    return __builtin_amdgcn_wmma_f32_16x16x32_bf16(
        /*neg_a=*/false, a.v, /*neg_b=*/false, b.v,
        /*c_mod=*/(short)0, c, /*reuse_a=*/false, /*reuse_b=*/false);
}

__device__ inline float wsum(float x) {
#pragma unroll
    for (int o = 16; o > 0; o >>= 1) x += __shfl_xor(x, o, 32);
    return x;
}

#if USE_TDM
// Issue TDM load of a 2D bf16 tile: 128 rows x 32 cols, row stride = dim0 elements.
// D# group0: count=1, lds_addr, 57-bit global byte address, type=2 ("image").
// D# group1: data_size=1 (2B), tensor_dim0=dim0, tensor_dim1=128,
//            tile_dim0=32, tile_dim1=128, tensor_dim0_stride=dim0.
__device__ inline void tdm_tile_128x32(unsigned lds_off, unsigned long long gaddr, int dim0) {
    u32x4 g0 = { 1u,
                 lds_off,
                 (unsigned)gaddr,
                 (unsigned)((gaddr >> 32) & 0x1FFFFFFull) | 0x80000000u };
    i32x8 g1 = { 0x00010000,                                        // data_size=2B
                 (int)(((unsigned)dim0 & 0xFFFFu) << 16),           // tensor_dim0[15:0]
                 (int)((((unsigned)dim0 >> 16) & 0xFFFFu) | (128u << 16)), // dim0[31:16] | tensor_dim1=128
                 (int)(32u << 16),                                  // tile_dim0=32
                 128,                                               // tile_dim1=128
                 dim0,                                              // tensor_dim0_stride[31:0]
                 0, 0 };
    i32x4 gz = {0, 0, 0, 0};
#if __clang_major__ >= 23
    i32x8 gz8 = {0, 0, 0, 0, 0, 0, 0, 0};
    __builtin_amdgcn_tensor_load_to_lds(g0, g1, gz, gz, gz8, 0);
#else
    __builtin_amdgcn_tensor_load_to_lds(g0, g1, gz, gz, 0);
#endif
}
__device__ inline void tdm_wait0() { __builtin_amdgcn_s_wait_tensorcnt(0); }
#endif

// ---------------- prep kernels ----------------
__global__ void k_f32_to_bf16(const float* in, unsigned short* out, long n) {
    long i = (long)blockIdx.x * blockDim.x + threadIdx.x;
    if (i < n) out[i] = f2bf(in[i]);
}

// in [v][R][C] f32 -> out [v][C][R] bf16
__global__ void k_transpose_bf16(const float* in, unsigned short* out, int R, int C) {
    int v = blockIdx.z;
    int idx = blockIdx.x * blockDim.x + threadIdx.x;
    if (idx >= R * C) return;
    int r = idx / C, c = idx % C;
    size_t b = (size_t)v * R * C;
    out[b + (size_t)c * R + r] = f2bf(in[b + (size_t)r * C + c]);
}

// ---------------- generic bf16 WMMA GEMM: out[v][n][128] = A[v][n][K] @ BT[v][128][K]^T ----------
// 4 waves/block, 16 rows/wave. B tile (128x32) staged once per block via TDM, double-buffered.
__global__ void k_gemm_bf16(const unsigned short* __restrict__ A,
                            const unsigned short* __restrict__ BT,
                            const float* __restrict__ bias,
                            float* __restrict__ out,
                            int N, int K, int relu) {
    int v = blockIdx.z;
    int wave = threadIdx.x >> 5, lane = threadIdx.x & 31;
    int rowbase = (blockIdx.x * (blockDim.x >> 5) + wave) * 16;
    int rA = lane & 15, kh = lane >> 4;

    const unsigned short* Arow = A + ((size_t)v * N + rowbase + rA) * K + 8 * kh;

    v8f zz = {0.f, 0.f, 0.f, 0.f, 0.f, 0.f, 0.f, 0.f};
    v8f acc[8];
#pragma unroll
    for (int t = 0; t < 8; t++) acc[t] = zz;

#if USE_TDM
    __shared__ __align__(16) unsigned short Btile[2][128][32];
    const unsigned short* Bg = BT + (size_t)v * H_ * K;
    if (wave == 0)
        tdm_tile_128x32((unsigned)(unsigned long long)(void*)&Btile[0][0][0],
                        (unsigned long long)(const void*)Bg, K);
    for (int k0 = 0; k0 < K; k0 += 32) {
        int s = (k0 >> 5) & 1;
        BfVec a = load16(Arow + k0);                 // overlap A fetch with TDM
        if (wave == 0) tdm_wait0();
        __syncthreads();                             // Btile[s] ready for all waves
        if (wave == 0 && k0 + 32 < K)
            tdm_tile_128x32((unsigned)(unsigned long long)(void*)&Btile[s ^ 1][0][0],
                            (unsigned long long)(const void*)(Bg + k0 + 32), K);
#pragma unroll
        for (int t = 0; t < 8; t++) {
            BfVec b = load16(&Btile[s][16 * t + rA][8 * kh]);
            acc[t] = wmma_bf16(a, b, acc[t]);
        }
        __syncthreads();                             // all reads done before next overwrite
    }
#else
    const unsigned short* Bbase = BT + (size_t)v * H_ * K + (size_t)rA * K + 8 * kh;
    for (int k0 = 0; k0 < K; k0 += 32) {
        BfVec a = load16(Arow + k0);
        BfVec b[8];
#pragma unroll
        for (int t = 0; t < 8; t++) b[t] = load16(Bbase + (size_t)(16 * t) * K + k0);
#pragma unroll
        for (int t = 0; t < 8; t++) acc[t] = wmma_bf16(a, b[t], acc[t]);
    }
#endif

#pragma unroll
    for (int t = 0; t < 8; t++) {
        int col = 16 * t + rA;
        float bv = bias ? bias[v * H_ + col] : 0.f;
#pragma unroll
        for (int r = 0; r < 8; r++) {
            int M = r + 8 * kh;                      // C-layout row
            float val = acc[t][r] + bv;
            if (relu) val = fmaxf(val, 0.f);
            out[((size_t)v * N + rowbase + M) * H_ + col] = val;
        }
    }
}

// ---------------- e1/e2 dots ----------------
__global__ void k_e1e2(const float* __restrict__ h, const float* __restrict__ a1,
                       const float* __restrict__ a2, float* e1, float* e2, int N) {
    int v = blockIdx.z;
    int wave = threadIdx.x >> 5, lane = threadIdx.x & 31;
    int node = blockIdx.x * (blockDim.x >> 5) + wave;
    if (node >= N) return;
    size_t base = ((size_t)v * N + node) * H_;
    float s1 = 0.f, s2 = 0.f;
#pragma unroll
    for (int j = 0; j < 4; j++) {
        int i = lane + 32 * j;
        float hv = h[base + i];
        s1 += hv * a1[v * H_ + i];
        s2 += hv * a2[v * H_ + i];
    }
    s1 = wsum(s1); s2 = wsum(s2);
    if (lane == 0) { e1[(size_t)v * N + node] = s1; e2[(size_t)v * N + node] = s2; }
}

// ---------------- fused masked-softmax @ h (flash-style, ones-matrix denominator) ----------------
__global__ void k_attn(const unsigned short* __restrict__ hT,   // [V][128][N] bf16
                       const float* __restrict__ e1, const float* __restrict__ e2,
                       const int* __restrict__ adj,             // [V][N][N]
                       const float* __restrict__ gac_b,         // [V][128]
                       const float* __restrict__ mlpout,        // [V][N][128] (incl. mlp_b)
                       unsigned short* __restrict__ featBf,     // [V][N][128] bf16
                       int N) {
    __shared__ __align__(16) unsigned short pbuf[4][16][32];
#if USE_TDM
    __shared__ __align__(16) unsigned short Btile[2][128][32];
#endif
    int v = blockIdx.z;
    int wave = threadIdx.x >> 5, lane = threadIdx.x & 31;
    int rowbase = (blockIdx.x * 4 + wave) * 16;
    int rA = lane & 15, kh = lane >> 4;

    float e1r[16];
#pragma unroll
    for (int j = 0; j < 16; j++) e1r[j] = e1[(size_t)v * N + rowbase + j];

    const float* e2p = e2 + (size_t)v * N;
    const int* adjp = adj + ((size_t)v * N + rowbase) * N;
    const unsigned short* hTv = hT + (size_t)v * H_ * N;
#if !USE_TDM
    const unsigned short* hTb = hTv + (size_t)rA * N + 8 * kh;
#endif

    BfVec ones;
    { uint4 o4; o4.x = o4.y = o4.z = o4.w = 0x3F803F80u; ones.q[0] = o4; ones.q[1] = o4; }

    v8f zz = {0.f, 0.f, 0.f, 0.f, 0.f, 0.f, 0.f, 0.f};
    v8f acc[8]; v8f accd = zz;
#pragma unroll
    for (int t = 0; t < 8; t++) acc[t] = zz;

#if USE_TDM
    if (wave == 0)
        tdm_tile_128x32((unsigned)(unsigned long long)(void*)&Btile[0][0][0],
                        (unsigned long long)(const void*)hTv, N);
#endif

    for (int m0 = 0; m0 < N; m0 += 32) {
        float e2v = e2p[m0 + lane];
        const int* ap = adjp + m0 + lane;
        bool pf = (m0 + 512 + lane) < N;             // prefetch adj ~16 tiles ahead
#pragma unroll
        for (int j = 0; j < 16; j++) {
            float e = e1r[j] + e2v;
            e = (e >= 0.f) ? e : 0.25f * e;          // leaky_relu(0.25)
            float p = (ap[(size_t)j * N] > 0) ? __expf(e) : 0.f;   // masked: exp(NEG)=0
            pbuf[wave][j][lane] = f2bf(p);
            if (pf) __builtin_prefetch(ap + (size_t)j * N + 512, 0, 0);
        }
#if USE_TDM
        int s = (m0 >> 5) & 1;
        if (wave == 0) tdm_wait0();
        __syncthreads();                             // pbuf + Btile[s] ready
        if (wave == 0 && m0 + 32 < N)
            tdm_tile_128x32((unsigned)(unsigned long long)(void*)&Btile[s ^ 1][0][0],
                            (unsigned long long)(const void*)(hTv + m0 + 32), N);
        BfVec a = load16(&pbuf[wave][rA][8 * kh]);
#pragma unroll
        for (int t = 0; t < 8; t++) {
            BfVec b = load16(&Btile[s][16 * t + rA][8 * kh]);
            acc[t] = wmma_bf16(a, b, acc[t]);
        }
#else
        __syncthreads();
        BfVec a = load16(&pbuf[wave][rA][8 * kh]);
#pragma unroll
        for (int t = 0; t < 8; t++) {
            BfVec b = load16(hTb + (size_t)(16 * t) * N + m0);
            acc[t] = wmma_bf16(a, b, acc[t]);
        }
#endif
        accd = wmma_bf16(a, ones, accd);             // row-sum denominator
        __syncthreads();
    }

#pragma unroll
    for (int t = 0; t < 8; t++) {
        int col = 16 * t + rA;
        float gb = gac_b[v * H_ + col];
#pragma unroll
        for (int r = 0; r < 8; r++) {
            int R = rowbase + r + 8 * kh;
            float val = acc[t][r] / accd[r] + gb;    // same (r,lane) -> same row
            val = (val >= 0.f) ? val : 0.25f * val;
            val += mlpout[((size_t)v * N + R) * H_ + col];
            featBf[((size_t)v * N + R) * H_ + col] = f2bf(val);
        }
    }
}

// ---------------- per-node HxH softmax attention ----------------
__global__ void k_sgfe(const float* __restrict__ Q, const float* __restrict__ Kf,
                       const float* __restrict__ Vf, unsigned short* __restrict__ aggBf, int N) {
    int v = blockIdx.z;
    int wave = threadIdx.x >> 5, lane = threadIdx.x & 31;
    int node = blockIdx.x * (blockDim.x >> 5) + wave;
    if (node >= N) return;
    size_t base = ((size_t)v * N + node) * H_;
    float q[4], k[4], vv[4];
#pragma unroll
    for (int j = 0; j < 4; j++) {
        int i = lane + 32 * j;
        q[j] = Q[base + i]; k[j] = Kf[base + i]; vv[j] = Vf[base + i];
    }
    float mk = fmaxf(fmaxf(k[0], k[1]), fmaxf(k[2], k[3]));
    float mn = fminf(fminf(k[0], k[1]), fminf(k[2], k[3]));
#pragma unroll
    for (int o = 16; o > 0; o >>= 1) {
        mk = fmaxf(mk, __shfl_xor(mk, o, 32));
        mn = fminf(mn, __shfl_xor(mn, o, 32));
    }
    const float inv = 0.08838834764831845f;          // 1/sqrt(128)
#pragma unroll
    for (int j = 0; j < 4; j++) {
        float qreg = q[j];
        for (int ii = 0; ii < 32; ii++) {
            float Qi = __shfl(qreg, ii, 32);
            float m = ((Qi >= 0.f) ? Qi * mk : Qi * mn) * inv;   // exact row max
            float sw = 0.f, swv = 0.f;
#pragma unroll
            for (int t = 0; t < 4; t++) {
                float w = __expf(Qi * k[t] * inv - m);
                sw += w; swv += w * vv[t];
            }
            sw = wsum(sw); swv = wsum(swv);
            if (lane == ii) aggBf[base + j * 32 + ii] = f2bf(swv / sw);
        }
    }
}

// ---------------- confidence gate + pack [N][V*H] ----------------
__global__ void k_gate(const float* __restrict__ feat2, const float* __restrict__ conf_w,
                       const float* __restrict__ conf_b, float* __restrict__ mm, int N) {
    int v = blockIdx.z;
    int wave = threadIdx.x >> 5, lane = threadIdx.x & 31;
    int node = blockIdx.x * (blockDim.x >> 5) + wave;
    if (node >= N) return;
    size_t base = ((size_t)v * N + node) * H_;
    float s = 0.f, f[4];
#pragma unroll
    for (int j = 0; j < 4; j++) {
        int i = lane + 32 * j;
        f[j] = feat2[base + i];
        s += f[j] * conf_w[v * H_ + i];
    }
    s = wsum(s);
    float conf = s + conf_b[v];
#pragma unroll
    for (int j = 0; j < 4; j++) {
        int i = lane + 32 * j;
        mm[(size_t)node * (V_ * H_) + v * H_ + i] = f[j] * conf;
    }
}

// ---------------- final classifier [N][384] @ [384][5] ----------------
__global__ void k_cls(const float* __restrict__ mm, const float* __restrict__ mm_w,
                      const float* __restrict__ mm_b, float* __restrict__ out, int N) {
    int wave = threadIdx.x >> 5, lane = threadIdx.x & 31;
    int node = blockIdx.x * (blockDim.x >> 5) + wave;
    if (node >= N) return;
    float acc[C_];
#pragma unroll
    for (int c = 0; c < C_; c++) acc[c] = 0.f;
#pragma unroll
    for (int j = 0; j < 12; j++) {
        int vh = lane + 32 * j;
        float x = mm[(size_t)node * (V_ * H_) + vh];
#pragma unroll
        for (int c = 0; c < C_; c++) acc[c] += x * mm_w[vh * C_ + c];
    }
#pragma unroll
    for (int c = 0; c < C_; c++) acc[c] = wsum(acc[c]);
    if (lane == 0) {
#pragma unroll
        for (int c = 0; c < C_; c++) out[(size_t)node * C_ + c] = acc[c] + mm_b[c];
    }
}

// ---------------- launch ----------------
extern "C" void kernel_launch(void* const* d_in, const int* in_sizes, int n_in,
                              void* d_out, int out_size, void* d_ws, size_t ws_size,
                              hipStream_t stream) {
    (void)in_sizes; (void)n_in; (void)out_size; (void)ws_size;
    const float* data   = (const float*)d_in[0];
    const int*   adj    = (const int*)  d_in[1];
    const float* gac_w  = (const float*)d_in[2];
    const float* gac_b  = (const float*)d_in[3];
    const float* a1     = (const float*)d_in[4];
    const float* a2     = (const float*)d_in[5];
    const float* mlp_w  = (const float*)d_in[6];
    const float* mlp_b  = (const float*)d_in[7];
    const float* q_w    = (const float*)d_in[8];
    const float* q_b    = (const float*)d_in[9];
    const float* k_w    = (const float*)d_in[10];
    const float* k_b    = (const float*)d_in[11];
    const float* v_w    = (const float*)d_in[12];
    const float* v_b    = (const float*)d_in[13];
    const float* fc_w   = (const float*)d_in[14];
    const float* fc_b   = (const float*)d_in[15];
    const float* conf_w = (const float*)d_in[16];
    const float* conf_b = (const float*)d_in[17];
    const float* mm_w   = (const float*)d_in[18];
    const float* mm_b   = (const float*)d_in[19];
    float* out = (float*)d_out;

    char* ws = (char*)d_ws;
    size_t off = 0;
    auto alloc = [&](size_t b) -> void* {
        void* p = ws + off;
        off = (off + b + 255) & ~(size_t)255;
        return p;
    };
    unsigned short* dataBf = (unsigned short*)alloc((size_t)V_ * N_ * D_ * 2);
    unsigned short* gacWT  = (unsigned short*)alloc((size_t)V_ * H_ * D_ * 2);
    unsigned short* mlpWT  = (unsigned short*)alloc((size_t)V_ * H_ * D_ * 2);
    unsigned short* qWT    = (unsigned short*)alloc((size_t)V_ * H_ * H_ * 2);
    unsigned short* kWT    = (unsigned short*)alloc((size_t)V_ * H_ * H_ * 2);
    unsigned short* vWT    = (unsigned short*)alloc((size_t)V_ * H_ * H_ * 2);
    unsigned short* fcWT   = (unsigned short*)alloc((size_t)V_ * H_ * H_ * 2);
    float*          h      = (float*)alloc((size_t)V_ * N_ * H_ * 4);
    float*          mlpout = (float*)alloc((size_t)V_ * N_ * H_ * 4);
    float*          e1     = (float*)alloc((size_t)V_ * N_ * 4);
    float*          e2     = (float*)alloc((size_t)V_ * N_ * 4);
    unsigned short* hT     = (unsigned short*)alloc((size_t)V_ * H_ * N_ * 2);
    unsigned short* featBf = (unsigned short*)alloc((size_t)V_ * N_ * H_ * 2);
    float*          Qb     = (float*)alloc((size_t)V_ * N_ * H_ * 4);
    float*          Kb     = (float*)alloc((size_t)V_ * N_ * H_ * 4);
    float*          Vb     = (float*)alloc((size_t)V_ * N_ * H_ * 4);
    unsigned short* aggBf  = (unsigned short*)alloc((size_t)V_ * N_ * H_ * 2);
    float*          feat2  = (float*)alloc((size_t)V_ * N_ * H_ * 4);
    float*          mmbuf  = (float*)alloc((size_t)N_ * V_ * H_ * 4);

    // prep: bf16 data + transposed bf16 weights
    long ntot = (long)V_ * N_ * D_;
    k_f32_to_bf16<<<dim3((unsigned)((ntot + 255) / 256)), 256, 0, stream>>>(data, dataBf, ntot);
    k_transpose_bf16<<<dim3((D_ * H_ + 255) / 256, 1, V_), 256, 0, stream>>>(gac_w, gacWT, D_, H_);
    k_transpose_bf16<<<dim3((D_ * H_ + 255) / 256, 1, V_), 256, 0, stream>>>(mlp_w, mlpWT, D_, H_);
    k_transpose_bf16<<<dim3((H_ * H_ + 255) / 256, 1, V_), 256, 0, stream>>>(q_w, qWT, H_, H_);
    k_transpose_bf16<<<dim3((H_ * H_ + 255) / 256, 1, V_), 256, 0, stream>>>(k_w, kWT, H_, H_);
    k_transpose_bf16<<<dim3((H_ * H_ + 255) / 256, 1, V_), 256, 0, stream>>>(v_w, vWT, H_, H_);
    k_transpose_bf16<<<dim3((H_ * H_ + 255) / 256, 1, V_), 256, 0, stream>>>(fc_w, fcWT, H_, H_);

    dim3 gg(N_ / 64, 1, V_);   // 4 waves/block, 16 rows/wave
    // h = data @ gac_w ; mlpout = data @ mlp_w + mlp_b
    k_gemm_bf16<<<gg, 128, 0, stream>>>(dataBf, gacWT, nullptr, h, N_, D_, 0);
    k_gemm_bf16<<<gg, 128, 0, stream>>>(dataBf, mlpWT, mlp_b, mlpout, N_, D_, 0);
    k_e1e2<<<dim3(N_ / 8, 1, V_), 256, 0, stream>>>(h, a1, a2, e1, e2, N_);
    k_transpose_bf16<<<dim3((N_ * H_ + 255) / 256, 1, V_), 256, 0, stream>>>(h, hT, N_, H_);
    // fused masked softmax + att @ h + gac_b + leaky + mlp residual
    k_attn<<<gg, 128, 0, stream>>>(hT, e1, e2, adj, gac_b, mlpout, featBf, N_);
    // Q/K/V projections
    k_gemm_bf16<<<gg, 128, 0, stream>>>(featBf, qWT, q_b, Qb, N_, H_, 0);
    k_gemm_bf16<<<gg, 128, 0, stream>>>(featBf, kWT, k_b, Kb, N_, H_, 0);
    k_gemm_bf16<<<gg, 128, 0, stream>>>(featBf, vWT, v_b, Vb, N_, H_, 0);
    // per-node HxH attention
    k_sgfe<<<dim3(N_ / 8, 1, V_), 256, 0, stream>>>(Qb, Kb, Vb, aggBf, N_);
    // fc + relu
    k_gemm_bf16<<<gg, 128, 0, stream>>>(aggBf, fcWT, fc_b, feat2, N_, H_, 1);
    // confidence gate + pack + classifier
    k_gate<<<dim3(N_ / 8, 1, V_), 256, 0, stream>>>(feat2, conf_w, conf_b, mmbuf, N_);
    k_cls<<<dim3(N_ / 8), 256, 0, stream>>>(mmbuf, mm_w, mm_b, out, N_);
}
